// AKT_39032662786545
// MI455X (gfx1250) — matrix operand order repaired and checked by reading, not compiled, not run on hardware
//
#include <hip/hip_runtime.h>

// Problem sizes (fixed by the reference)
#define Bsz   32
#define Ssz   512
#define Dsz   512
#define Hh    8
#define DFFsz 2048
#define DKsz  64
#define NEGV  (-1e32f)

typedef __attribute__((ext_vector_type(16))) __bf16       bf16x16;
typedef __attribute__((ext_vector_type(8)))  float        f32x8;
typedef __attribute__((ext_vector_type(4)))  float        f32x4;
typedef __attribute__((ext_vector_type(4)))  unsigned int u32x4;
typedef __attribute__((ext_vector_type(2)))  unsigned int u32x2;

union Frag {
    bf16x16        v;
    unsigned int   w[8];
    u32x4          q[2];
};

// Pack two f32 -> one dword holding (hi_bf16 << 16) | lo_bf16, RTNE.
__device__ __forceinline__ unsigned int pk_bf16(float lo, float hi) {
#if __has_builtin(__builtin_amdgcn_cvt_pk_bf16_f32)
    auto r = __builtin_amdgcn_cvt_pk_bf16_f32(lo, hi);
    return __builtin_bit_cast(unsigned int, r);
#else
    union { __bf16 h[2]; unsigned int w; } u;
    u.h[0] = (__bf16)lo;
    u.h[1] = (__bf16)hi;
    return u.w;
#endif
}

__device__ __forceinline__ u32x4 pk4(f32x4 a, f32x4 b) {
    u32x4 o;
    o.x = pk_bf16(a.x, a.y); o.y = pk_bf16(a.z, a.w);
    o.z = pk_bf16(b.x, b.y); o.w = pk_bf16(b.z, b.w);
    return o;
}

__device__ __forceinline__ f32x8 wmma_bf16(const Frag& a, const Frag& b, f32x8 c) {
    return __builtin_amdgcn_wmma_f32_16x16x32_bf16(
        /*neg_a=*/false, a.v, /*neg_b=*/false, b.v,
        /*c_mod=*/(short)0, c, /*reuse_a=*/false, /*reuse_b=*/false);
}

__device__ __forceinline__ float wredmax(float x) {
    #pragma unroll
    for (int o = 16; o > 0; o >>= 1) x = fmaxf(x, __shfl_xor(x, o, 32));
    return x;
}
__device__ __forceinline__ float wredsum(float x) {
    #pragma unroll
    for (int o = 16; o > 0; o >>= 1) x += __shfl_xor(x, o, 32);
    return x;
}
__device__ __forceinline__ float wscan_incl(float x) {
    const int lane = threadIdx.x & 31;
    #pragma unroll
    for (int o = 1; o < 32; o <<= 1) {
        float n = __shfl_up(x, o, 32);
        if (lane >= o) x += n;
    }
    return x;
}

// ---------------------------------------------------------------------------
// GEMM: C[M,N] = act(A[M,K] @ W[K,N] + bias[N]); N,K compile-time so all
// address math folds to immediate offsets. fp32 memory, bf16 WMMA, f32 accum.
// Block = 128x64 tile, 256 threads (8 waves in 4x2), wave owns 32x32 quadrant.
// ---------------------------------------------------------------------------
template<int N, int K, int RELU>
__global__ __launch_bounds__(256)
void k_gemm(const float* __restrict__ A, const float* __restrict__ W,
            const float* __restrict__ bias, float* __restrict__ C) {
    __shared__ unsigned short As[128][40];  // rows padded to 80B (16B aligned)
    __shared__ unsigned short Bs[64][40];   // W tile, stored transposed [n][k]

    const int tid  = threadIdx.x;
    const int wave = tid >> 5, lane = tid & 31;
    const int lr   = lane & 15, hi = lane >> 4;
    const int m0   = blockIdx.x * 128, n0 = blockIdx.y * 64;
    const int wm   = (wave & 3) * 32, wn = (wave >> 2) * 32;

    f32x8 acc[2][2] = {};

    for (int k0 = 0; k0 < K; k0 += 32) {
        if (k0 + 32 < K) {   // next tiles -> global_prefetch_b8
            __builtin_prefetch(A + (size_t)(m0 + (tid >> 1)) * K + k0 + 32, 0, 1);
            __builtin_prefetch(W + (size_t)(k0 + 32 + (tid >> 3)) * N + n0, 0, 1);
        }
        __syncthreads();
        // A tile 128x32: packed along k -> b64 LDS stores
        #pragma unroll
        for (int it = 0; it < 4; ++it) {
            int idx = tid + it * 256;                 // 0..1023 f32x4 units
            int r = idx >> 3, c = (idx & 7) << 2;
            f32x4 fa = *(const f32x4*)(A + (size_t)(m0 + r) * K + k0 + c);
            u32x2 o; o.x = pk_bf16(fa.x, fa.y); o.y = pk_bf16(fa.z, fa.w);
            *(u32x2*)&As[r][c] = o;
        }
        // W tile 32x64 -> transposed [n][k]: load two adjacent k rows
        // (coalesced along n), pack the k-pair -> b32 LDS stores
        {
            int kr = (tid >> 4) << 1;                 // even k row 0..30
            int nc = (tid & 15) << 2;                 // 0..60
            const float* wp = W + (size_t)(k0 + kr) * N + n0 + nc;
            f32x4 g0 = *(const f32x4*)wp;
            f32x4 g1 = *(const f32x4*)(wp + N);
            *(unsigned int*)&Bs[nc + 0][kr] = pk_bf16(g0.x, g1.x);
            *(unsigned int*)&Bs[nc + 1][kr] = pk_bf16(g0.y, g1.y);
            *(unsigned int*)&Bs[nc + 2][kr] = pk_bf16(g0.z, g1.z);
            *(unsigned int*)&Bs[nc + 3][kr] = pk_bf16(g0.w, g1.w);
        }
        __syncthreads();

        Frag a[2], b[2];
        #pragma unroll
        for (int mt = 0; mt < 2; ++mt) {   // A frag: lane<16 K{0..7,16..23}
            const u32x4* p = (const u32x4*)&As[wm + mt * 16 + lr][0];
            a[mt].q[0] = p[hi];
            a[mt].q[1] = p[hi + 2];
        }
        #pragma unroll
        for (int nt = 0; nt < 2; ++nt) {   // B frag: lane<16 K{0..7,16..23}
            const u32x4* p = (const u32x4*)&Bs[wn + nt * 16 + lr][0];
            b[nt].q[0] = p[hi];
            b[nt].q[1] = p[hi + 2];
        }
        #pragma unroll
        for (int mt = 0; mt < 2; ++mt)
            #pragma unroll
            for (int nt = 0; nt < 2; ++nt)
                acc[mt][nt] = wmma_bf16(a[mt], b[nt], acc[mt][nt]);
    }

    // Epilogue: N compile-time -> immediate store offsets
    float* cbase = C + (size_t)(m0 + wm + hi * 8) * N + (n0 + wn + lr);
    #pragma unroll
    for (int mt = 0; mt < 2; ++mt)
        #pragma unroll
        for (int nt = 0; nt < 2; ++nt) {
            float bv = bias[n0 + wn + nt * 16 + lr];
            float* q = cbase + (size_t)mt * 16 * N + nt * 16;
            #pragma unroll
            for (int r = 0; r < 8; ++r) {   // C layout: M=r+8*hi, N=lane%16
                float vv = acc[mt][nt][r] + bv;
                if (RELU) vv = fmaxf(vv, 0.f);
                q[(size_t)r * N] = vv;
            }
        }
}

// ---------------------------------------------------------------------------
// Fused attention for one (b, h, 32-row block):
//   scores = QK^T/8 (WMMA, LDS-staged bf16) -> f32 score slab in LDS ->
//   masked softmax -> cumsum -> distance-decay rescale -> second softmax
//   (probs written back to LDS as packed bf16) -> P@V (WMMA, pure b128 loads).
// 256 threads = 8 waves. Score slab never touches HBM.
// LDS plan (97.5KB peak):
//   [0, 66560)        sc  f32 [32][520]   phases 1-2   | Vt bf16 [64][40] ph.3
//   [66560, 99840)    Ps bf16 [32][520]   phases 2-3   | Qs/Ks bf16   ph.1
// ---------------------------------------------------------------------------
#define ROWS 32
#define SCS  520   // padded row stride for bank spread (f32 and bf16 slabs)

__global__ __launch_bounds__(256)
void k_attn(const float* __restrict__ qp, const float* __restrict__ kp,
            const float* __restrict__ vp, const float* __restrict__ gammas,
            int layer, int mdiag, int zero_pad, float* __restrict__ out) {
    __shared__ __align__(16) unsigned char smem[32 * SCS * 4 + 32 * SCS * 2];

    float*          sc = (float*)smem;                            // [32][520]
    unsigned short* Ps = (unsigned short*)(smem + 32 * SCS * 4);  // [32][520]
    unsigned short* Qs = Ps;                                      // [32][72]
    unsigned short* Ks = Ps + 32 * 72;                            // [128][72]
    unsigned short* Vt = (unsigned short*)smem;                   // [64][40]

    const int tid  = threadIdx.x;
    const int wave = tid >> 5, lane = tid & 31;
    const int lr   = lane & 15, hi = lane >> 4;
    const int i0   = blockIdx.x * ROWS;
    const int h    = blockIdx.y, b = blockIdx.z;
    const size_t hb = (size_t)b * Ssz * Dsz + (size_t)h * DKsz;

    // -------- phase 1: scores via LDS-staged bf16 tiles -------------------
    {
        // stage Q block 32x64 (8 f32 per thread, one b128 store)
        {
            int r = tid >> 3, part = (tid & 7) << 3;
            const float* qb = qp + hb + (size_t)(i0 + r) * Dsz + part;
            f32x4 f0 = *(const f32x4*)qb;
            f32x4 f1 = *(const f32x4*)(qb + 4);
            *(u32x4*)&Qs[r * 72 + part] = pk4(f0, f1);
        }
        for (int jt = 0; jt < 4; ++jt) {       // 4 chunks of 128 keys
            __syncthreads();
            {   // stage K chunk 128x64 (32 f32 per thread, 4 b128 stores)
                int r = tid >> 1, half = (tid & 1) << 5;
                const float* kb = kp + hb + (size_t)(jt * 128 + r) * Dsz + half;
                #pragma unroll
                for (int u = 0; u < 4; ++u) {
                    f32x4 f0 = *(const f32x4*)(kb + u * 8);
                    f32x4 f1 = *(const f32x4*)(kb + u * 8 + 4);
                    *(u32x4*)&Ks[r * 72 + half + u * 8] = pk4(f0, f1);
                }
            }
            __syncthreads();
            // wave owns one 16-wide n-tile of this chunk, 2 m-tiles
            f32x8 acc[2] = {};
            #pragma unroll
            for (int ks = 0; ks < 2; ++ks) {   // DK=64 -> 2 WMMA k-steps
                const int kd = ks * 32;
                Frag a0, a1, bf;
                const u32x4* pq0 = (const u32x4*)&Qs[(lr)      * 72 + kd];
                const u32x4* pq1 = (const u32x4*)&Qs[(16 + lr) * 72 + kd];
                a0.q[0] = pq0[hi]; a0.q[1] = pq0[hi + 2];
                a1.q[0] = pq1[hi]; a1.q[1] = pq1[hi + 2];
                const u32x4* pk_ = (const u32x4*)&Ks[(wave * 16 + lr) * 72 + kd];
                bf.q[0] = pk_[hi]; bf.q[1] = pk_[hi + 2];
                acc[0] = wmma_bf16(a0, bf, acc[0]);
                acc[1] = wmma_bf16(a1, bf, acc[1]);
            }
            const int col = jt * 128 + wave * 16 + lr;
            #pragma unroll
            for (int mt = 0; mt < 2; ++mt)
                #pragma unroll
                for (int r = 0; r < 8; ++r)
                    sc[(mt * 16 + r + hi * 8) * SCS + col] = acc[mt][r] * 0.125f;
        }
    }
    __syncthreads();

    // -------- phase 2: per-row rescoring (wave owns 4 rows) ---------------
    {
        float g     = gammas[layer * Hh + h];
        float gamma = -((g > 20.f) ? g : log1pf(__expf(g)));   // -softplus
        #pragma unroll
        for (int rr = 0; rr < 4; ++rr) {
            const int row = wave * 4 + rr;
            const int ri  = i0 + row;
            const int jb  = lane * 16;             // each lane: 16 contiguous j
            float v[16]; int val[16];
            #pragma unroll
            for (int t4 = 0; t4 < 4; ++t4) {
                f32x4 f = *(const f32x4*)&sc[row * SCS + jb + t4 * 4];
                v[t4*4+0] = f.x; v[t4*4+1] = f.y; v[t4*4+2] = f.z; v[t4*4+3] = f.w;
            }
            #pragma unroll
            for (int t = 0; t < 16; ++t) val[t] = (jb + t) <= (ri + mdiag);

            // softmax #1 (masked), then * mask
            float mx = NEGV;
            #pragma unroll
            for (int t = 0; t < 16; ++t) mx = fmaxf(mx, val[t] ? v[t] : NEGV);
            mx = wredmax(mx);
            float p[16], s = 0.f;
            #pragma unroll
            for (int t = 0; t < 16; ++t) { float e = __expf((val[t] ? v[t] : NEGV) - mx); p[t] = e; s += e; }
            s = wredsum(s);
            float inv = 1.f / s;
            #pragma unroll
            for (int t = 0; t < 16; ++t) p[t] = val[t] ? p[t] * inv : 0.f;

            // inclusive cumsum along j (lane-local scan + wave scan of totals)
            float run = 0.f, cs[16];
            #pragma unroll
            for (int t = 0; t < 16; ++t) { run += p[t]; cs[t] = run; }
            float incl = wscan_incl(run);
            float excl = incl - run;
            float dtot = __shfl(incl, 31, 32);     // disttotal = row sum

            // distance-decay rescale
            float s2[16];
            #pragma unroll
            for (int t = 0; t < 16; ++t) {
                float dc  = excl + cs[t];
                float pos = fabsf((float)(ri - (jb + t)));
                float d   = sqrtf(fmaxf((dtot - dc) * pos, 0.f));
                float eff = __expf(d * gamma);
                eff  = fminf(fmaxf(eff, 1e-5f), 1e5f);
                s2[t] = val[t] ? v[t] * eff : NEGV;
            }

            // softmax #2
            float mx2 = NEGV;
            #pragma unroll
            for (int t = 0; t < 16; ++t) mx2 = fmaxf(mx2, s2[t]);
            mx2 = wredmax(mx2);
            float z2 = 0.f, q2[16];
            #pragma unroll
            for (int t = 0; t < 16; ++t) { float e = __expf(s2[t] - mx2); q2[t] = e; z2 += e; }
            z2 = wredsum(z2);
            float keep = (zero_pad && ri == 0) ? 0.f : (1.f / z2);

            // write final probabilities as packed bf16 (feeds P@V directly)
            u32x4 o0, o1;
            o0.x = pk_bf16(q2[0]*keep,  q2[1]*keep);
            o0.y = pk_bf16(q2[2]*keep,  q2[3]*keep);
            o0.z = pk_bf16(q2[4]*keep,  q2[5]*keep);
            o0.w = pk_bf16(q2[6]*keep,  q2[7]*keep);
            o1.x = pk_bf16(q2[8]*keep,  q2[9]*keep);
            o1.y = pk_bf16(q2[10]*keep, q2[11]*keep);
            o1.z = pk_bf16(q2[12]*keep, q2[13]*keep);
            o1.w = pk_bf16(q2[14]*keep, q2[15]*keep);
            *(u32x4*)&Ps[row * SCS + jb]     = o0;
            *(u32x4*)&Ps[row * SCS + jb + 8] = o1;
        }
    }
    __syncthreads();

    // -------- phase 3: out = P @ V  (each wave owns one 16x16 tile) -------
    {
        const int mt = wave >> 2, nt = wave & 3;   // 2 m-tiles x 4 n-tiles
        f32x8 acc = {};
        for (int ks = 0; ks < 16; ++ks) {          // K = S = 512 -> 16 steps
            __syncthreads();
            const int kj = ks * 32;
            {   // stage V[kj..kj+31][0..63] transposed: load adjacent j-rows,
                // pack the pair -> b32 LDS stores
                int r = (tid >> 4) << 1, c = (tid & 15) << 2;
                const float* vb = vp + hb + (size_t)(kj + r) * Dsz + c;
                f32x4 f0 = *(const f32x4*)vb;
                f32x4 f1 = *(const f32x4*)(vb + Dsz);
                *(unsigned int*)&Vt[(c + 0) * 40 + r] = pk_bf16(f0.x, f1.x);
                *(unsigned int*)&Vt[(c + 1) * 40 + r] = pk_bf16(f0.y, f1.y);
                *(unsigned int*)&Vt[(c + 2) * 40 + r] = pk_bf16(f0.z, f1.z);
                *(unsigned int*)&Vt[(c + 3) * 40 + r] = pk_bf16(f0.w, f1.w);
            }
            __syncthreads();
            Frag a, bf;
            const u32x4* pa = (const u32x4*)&Ps[(mt * 16 + lr) * SCS + kj];
            a.q[0] = pa[hi];
            a.q[1] = pa[hi + 2];
            const u32x4* pb = (const u32x4*)&Vt[(nt * 16 + lr) * 40];
            bf.q[0] = pb[hi];
            bf.q[1] = pb[hi + 2];
            acc = wmma_bf16(a, bf, acc);
        }
        float* ob = out + ((size_t)b * Ssz + i0 + mt * 16 + hi * 8) * Dsz
                        + h * DKsz + nt * 16 + lr;
        #pragma unroll
        for (int r = 0; r < 8; ++r)
            ob[(size_t)r * Dsz] = acc[r];
    }
}

// ---------------------------------------------------------------------------
// out = LayerNorm(a + res) * g + b   (one 512-wide row per block)
// ---------------------------------------------------------------------------
__global__ __launch_bounds__(256)
void k_add_ln(const float* __restrict__ a, const float* __restrict__ res,
              const float* __restrict__ g, const float* __restrict__ bta,
              float* __restrict__ out) {
    const int    tid  = threadIdx.x;
    const size_t base = (size_t)blockIdx.x * Dsz;
    float x0 = a[base + tid]       + res[base + tid];
    float x1 = a[base + tid + 256] + res[base + tid + 256];
    float s  = x0 + x1, ss = x0 * x0 + x1 * x1;
    s = wredsum(s); ss = wredsum(ss);
    __shared__ float sh[16];
    const int wave = tid >> 5, lane = tid & 31;
    if (lane == 0) { sh[wave] = s; sh[8 + wave] = ss; }
    __syncthreads();
    s = 0.f; ss = 0.f;
    #pragma unroll
    for (int w = 0; w < 8; ++w) { s += sh[w]; ss += sh[8 + w]; }
    float mean = s * (1.f / 512.f);
    float var  = ss * (1.f / 512.f) - mean * mean;
    float rs   = rsqrtf(var + 1e-5f);
    out[base + tid]       = (x0 - mean) * rs * g[tid]       + bta[tid];
    out[base + tid + 256] = (x1 - mean) * rs * g[tid + 256] + bta[tid + 256];
}

// ---------------------------------------------------------------------------
template<int N, int K, int RELU>
static inline void launch_gemm(const float* A, const float* W, const float* bias,
                               float* C, hipStream_t stream) {
    dim3 grid((Bsz * Ssz) / 128, N / 64);
    hipLaunchKernelGGL((k_gemm<N, K, RELU>), grid, dim3(256), 0, stream, A, W, bias, C);
}

extern "C" void kernel_launch(void* const* d_in, const int* in_sizes, int n_in,
                              void* d_out, int out_size, void* d_ws, size_t ws_size,
                              hipStream_t stream) {
    const float* q_embed  = (const float*)d_in[0];
    const float* qa_embed = (const float*)d_in[1];
    const float* Wk  = (const float*)d_in[2];
    const float* bk  = (const float*)d_in[3];
    const float* Wv  = (const float*)d_in[4];
    const float* bv  = (const float*)d_in[5];
    const float* Wo  = (const float*)d_in[6];
    const float* bo  = (const float*)d_in[7];
    const float* gam = (const float*)d_in[8];
    const float* l1g = (const float*)d_in[9];
    const float* l1b = (const float*)d_in[10];
    const float* W1  = (const float*)d_in[11];
    const float* b1  = (const float*)d_in[12];
    const float* W2  = (const float*)d_in[13];
    const float* b2  = (const float*)d_in[14];
    const float* l2g = (const float*)d_in[15];
    const float* l2b = (const float*)d_in[16];

    const int    M  = Bsz * Ssz;               // 16384 rows
    const size_t SE = (size_t)M * Dsz;         // f32 elements per act buffer

    float* ws = (float*)d_ws;                  // 8 x 33.5MB arenas = 268MB
    float* qp = ws;
    float* kq = ws + 1 * SE;
    float* vv = ws + 2 * SE;
    float* a0 = ws + 3 * SE;
    float* ao = ws + 4 * SE;
    float* t1 = ws + 5 * SE;
    float* y1 = ws + 6 * SE;
    float* x1 = ws + 7 * SE;
    float* hh = qp;   // FFN hidden [M,DFF] aliases qp..a0 (dead during FFN)
    float* ff = ao;   // FFN output aliases ao (dead after ln1)

    auto layer = [&](int i, int mdiag, int zp, const float* q, const float* k,
                     const float* v, int apply_pos, float* outp) {
        const size_t wo = (size_t)i * Dsz * Dsz;
        launch_gemm<Dsz, Dsz, 0>(q, Wk + wo, bk + i * Dsz, qp, stream);  // q uses k_linear
        launch_gemm<Dsz, Dsz, 0>(k, Wk + wo, bk + i * Dsz, kq, stream);
        launch_gemm<Dsz, Dsz, 0>(v, Wv + wo, bv + i * Dsz, vv, stream);
        hipLaunchKernelGGL(k_attn, dim3(Ssz / ROWS, Hh, Bsz), dim3(256), 0, stream,
                           qp, kq, vv, gam, i, mdiag, zp, a0);
        launch_gemm<Dsz, Dsz, 0>(a0, Wo + wo, bo + i * Dsz, ao, stream);
        float* l1out = apply_pos ? t1 : outp;
        hipLaunchKernelGGL(k_add_ln, dim3(M), dim3(256), 0, stream,
                           ao, q, l1g + i * Dsz, l1b + i * Dsz, l1out);
        if (apply_pos) {
            launch_gemm<DFFsz, Dsz, 1>(t1, W1 + (size_t)i * Dsz * DFFsz, b1 + i * DFFsz, hh, stream);
            launch_gemm<Dsz, DFFsz, 0>(hh, W2 + (size_t)i * DFFsz * Dsz, b2 + i * Dsz, ff, stream);
            hipLaunchKernelGGL(k_add_ln, dim3(M), dim3(256), 0, stream,
                               ff, t1, l2g + i * Dsz, l2b + i * Dsz, outp);
        }
    };

    // blocks_1: self-attn on qa stream, causal incl. diagonal, with FFN
    layer(0,  0, 0, qa_embed, qa_embed, qa_embed, 1, y1);
    // blocks_2 layer A: self-attn on q stream, no FFN
    layer(1,  0, 0, q_embed,  q_embed,  q_embed,  0, x1);
    // blocks_2 layer B: strict-causal cross-attn (values = y stream), zero_pad, FFN
    layer(2, -1, 1, x1, x1, y1, 1, (float*)d_out);

    (void)in_sizes; (void)n_in; (void)out_size; (void)ws_size;
}
